// MatryoshkaSampledSoftmaxLoss_53300544143625
// MI455X (gfx1250) — matrix operand order repaired and checked by reading, not compiled, not run on hardware
//
#include <hip/hip_runtime.h>
#include <hip/hip_bf16.h>
#include <math.h>

#define V_SZ    50257
#define V_PAD   50304
#define D_SZ    1024
#define T_SZ    8192
#define CHUNK_SZ 128
#define NCHUNK  64
#define NCAND   2048
#define NSEG    8
#define SEGC    256
#define LOWR    32
#define KBLK    64
#define BSTR    (KBLK + 8)   // padded LDS stride (halfs) -> conflict-free b128
#define LSTR    (LOWR + 8)
#define NBIN    4096

typedef __attribute__((ext_vector_type(16))) _Float16 v16h;
typedef __attribute__((ext_vector_type(8)))  _Float16 v8h;
typedef __attribute__((ext_vector_type(8)))  float    v8f;

#if defined(__AMDGCN__) && __has_builtin(__builtin_amdgcn_global_load_async_to_lds_b128)
#define HAVE_ASYNC 1
#else
#define HAVE_ASYNC 0
#endif

#if HAVE_ASYNC
typedef int v4i __attribute__((vector_size(16)));
typedef __attribute__((address_space(1))) v4i* as1_v4i_p;
typedef __attribute__((address_space(3))) v4i* as3_v4i_p;
__device__ __forceinline__ void async_cp16(const _Float16* g, _Float16* l) {
    __builtin_amdgcn_global_load_async_to_lds_b128(
        (as1_v4i_p)g, (as3_v4i_p)l, 0, 0);
}
#endif

__device__ __forceinline__ void wait_async0() {
#if defined(__AMDGCN__)
#if __has_builtin(__builtin_amdgcn_s_wait_asynccnt)
    __builtin_amdgcn_s_wait_asynccnt(0);
#else
    asm volatile("s_wait_asynccnt 0x0" ::: "memory");
#endif
#endif
}

// ---------------- fragment loaders (per CDNA5 ISA layouts) -------
// A 16x32 f16: lane m=lane&15, half=lane>>4. halves [0..7]=p[0..7], [8..15]=p[16..23]
__device__ __forceinline__ v16h load_a(const _Float16* p) {
    v8h lo = *(const v8h*)p;
    v8h hi = *(const v8h*)(p + 16);
    v16h r;
#pragma unroll
    for (int i = 0; i < 8; ++i) { r[i] = lo[i]; r[i + 8] = hi[i]; }
    return r;
}
// B 32x16 f16: lane holds one column, halves [0..7]=p[0..7], [8..15]=p[16..23]
__device__ __forceinline__ v16h load_b(const _Float16* p) {
    v8h lo = *(const v8h*)p;
    v8h hi = *(const v8h*)(p + 16);
    v16h r;
#pragma unroll
    for (int i = 0; i < 8; ++i) { r[i] = lo[i]; r[i + 8] = hi[i]; }
    return r;
}

// ------- kernel A: embedding norms + pre-normalized f16 tables --------------
__global__ __launch_bounds__(256) void k_emb_norms(const float* __restrict__ emb,
                                                   float* __restrict__ invF,
                                                   float* __restrict__ invL,
                                                   _Float16* __restrict__ eF16,
                                                   _Float16* __restrict__ eL16) {
    int row = blockIdx.x * 8 + (threadIdx.x >> 5);
    int lane = threadIdx.x & 31;
    if (row >= V_SZ) return;
    const float* p = emb + (size_t)row * D_SZ;
    float xr[32];
    float ss = 0.f;
#pragma unroll
    for (int j = 0; j < 32; ++j) { xr[j] = p[lane + 32 * j]; ss += xr[j] * xr[j]; }
    float sl = xr[0] * xr[0];
#pragma unroll
    for (int off = 16; off; off >>= 1) {
        ss += __shfl_xor(ss, off, 32);
        sl += __shfl_xor(sl, off, 32);
    }
    float invf = 1.f / fmaxf(sqrtf(ss), 1e-12f);
    float invl = 1.f / fmaxf(sqrtf(sl), 1e-12f);
    if (lane == 0) { invF[row] = invf; invL[row] = invl; }
#pragma unroll
    for (int j = 0; j < 32; ++j)
        eF16[(size_t)row * D_SZ + lane + 32 * j] = (_Float16)(xr[j] * invf);
    eL16[(size_t)row * LOWR + lane] = (_Float16)(xr[0] * invl);
}

// ---------------- kernel B: h norms, f16 staging, chunk means ---------------
__global__ __launch_bounds__(256) void k_h_norms(const float* __restrict__ h,
                                                 float* __restrict__ invHF,
                                                 float* __restrict__ invHL,
                                                 _Float16* __restrict__ hF16,
                                                 _Float16* __restrict__ hL16,
                                                 float* __restrict__ hmean) {
    int row = blockIdx.x * 8 + (threadIdx.x >> 5);
    int lane = threadIdx.x & 31;
    if (row >= T_SZ) return;
    const float* p = h + (size_t)row * D_SZ;
    float xr[32];
    float ss = 0.f;
#pragma unroll
    for (int j = 0; j < 32; ++j) { xr[j] = p[lane + 32 * j]; ss += xr[j] * xr[j]; }
    float sl = xr[0] * xr[0];
#pragma unroll
    for (int off = 16; off; off >>= 1) {
        ss += __shfl_xor(ss, off, 32);
        sl += __shfl_xor(sl, off, 32);
    }
    float invf = 1.f / fmaxf(sqrtf(ss), 1e-12f);
    float invl = 1.f / fmaxf(sqrtf(sl), 1e-12f);
    if (lane == 0) { invHF[row] = invf; invHL[row] = invl; }
#pragma unroll
    for (int j = 0; j < 32; ++j)
        hF16[(size_t)row * D_SZ + lane + 32 * j] = (_Float16)(xr[j] * invf);
    float xl = xr[0] * invl;                  // element index == lane (<32)
    hL16[(size_t)row * LOWR + lane] = (_Float16)xl;
    atomicAdd(&hmean[(row >> 7) * LOWR + lane], xl);
}

// ------- kernel C1: candidate scan (vocab row read once, all 64 chunks) -----
__global__ __launch_bounds__(256) void k_scan(const _Float16* __restrict__ eL16,
                                              const float* __restrict__ hmean,
                                              float* __restrict__ scores) {
    __shared__ float hm[NCHUNK * LOWR];   // 8 KB
    for (int i = threadIdx.x; i < NCHUNK * LOWR; i += 256) hm[i] = hmean[i];
    __syncthreads();
    int v = blockIdx.x * blockDim.x + threadIdx.x;
    if (v >= V_SZ) return;
    float wv[LOWR];
    const _Float16* e = eL16 + (size_t)v * LOWR;
#pragma unroll
    for (int l = 0; l < LOWR; ++l) wv[l] = (float)e[l];
    for (int n = 0; n < NCHUNK; ++n) {
        float s = 0.f;
#pragma unroll
        for (int l = 0; l < LOWR; ++l) s += hm[n * LOWR + l] * wv[l];
        scores[(size_t)n * V_PAD + v] = s;   // top-k invariant to +scale
    }
}

// ---------------- kernel C2: per-chunk radix top-2048 -----------------------
__global__ __launch_bounds__(1024) void k_topk(const float* __restrict__ scores,
                                               int* __restrict__ topIdx) {
    __shared__ int hist[NBIN];
    __shared__ int s_thr, s_cntgt, c1, c2;
    int n = blockIdx.x;
    int tid = threadIdx.x;
    for (int b = tid; b < NBIN; b += 1024) hist[b] = 0;
    if (tid == 0) { c1 = 0; c2 = 0; }
    __syncthreads();
    const float* sc = scores + (size_t)n * V_PAD;
    for (int v = tid; v < V_SZ; v += 1024) {
        unsigned u = __float_as_uint(sc[v]);
        unsigned key = (u & 0x80000000u) ? ~u : (u | 0x80000000u);
        atomicAdd(&hist[key >> 20], 1);
    }
    __syncthreads();
    if (tid == 0) {
        int acc = 0, thr = 0, cg = 0;
        for (int b = NBIN - 1; b >= 0; --b) {
            if (acc + hist[b] >= NCAND) { thr = b; cg = acc; break; }
            acc += hist[b];
        }
        s_thr = thr; s_cntgt = cg;
    }
    __syncthreads();
    int thr = s_thr, cg = s_cntgt;
    for (int v = tid; v < V_SZ; v += 1024) {
        unsigned u = __float_as_uint(sc[v]);
        unsigned key = (u & 0x80000000u) ? ~u : (u | 0x80000000u);
        int b = (int)(key >> 20);
        if (b > thr) {
            int p0 = atomicAdd(&c1, 1);
            topIdx[n * NCAND + p0] = v;
        } else if (b == thr) {
            int q = atomicAdd(&c2, 1);
            if (cg + q < NCAND) topIdx[n * NCAND + cg + q] = v;
        }
    }
}

// ---------------- kernel E: target similarities -----------------------------
__global__ __launch_bounds__(256) void k_tgt(const float* __restrict__ h,
                                             const float* __restrict__ emb,
                                             const int* __restrict__ tgt,
                                             const float* __restrict__ invF,
                                             const float* __restrict__ invL,
                                             const float* __restrict__ invHF,
                                             const float* __restrict__ invHL,
                                             float* __restrict__ tF,
                                             float* __restrict__ tL) {
    int row = blockIdx.x * 8 + (threadIdx.x >> 5);
    int lane = threadIdx.x & 31;
    if (row >= T_SZ) return;
    int t = tgt[row];
    const float* p = h + (size_t)row * D_SZ;
    const float* e = emb + (size_t)t * D_SZ;
    float a0 = p[lane] * e[lane];
    float dl = a0, d = a0;
#pragma unroll
    for (int j = 1; j < 32; ++j) d += p[lane + 32 * j] * e[lane + 32 * j];
#pragma unroll
    for (int off = 16; off; off >>= 1) {
        d  += __shfl_xor(d, off, 32);
        dl += __shfl_xor(dl, off, 32);
    }
    if (lane == 0) {
        tF[row] = d * invHF[row] * invF[t];
        tL[row] = dl * invHL[row] * invL[t];
    }
}

// ---------------- masked partial log-sum-exp epilogue -----------------------
__device__ __forceinline__ void epilogue(const v8f* acc, const int* cidx,
                                         const int* tgtRow16, float scaleF,
                                         float2* out) {
    int lane = threadIdx.x & 31;
    int mrow = lane & 15, hf = lane >> 4;
    float m[8], s[8];
    int tg[8];
#pragma unroll
    for (int v = 0; v < 8; ++v) { m[v] = -INFINITY; s[v] = 0.f; tg[v] = tgtRow16[v + 8 * hf]; }
#pragma unroll
    for (int t = 0; t < 16; ++t) {
        int c = cidx[t * 16 + mrow];
#pragma unroll
        for (int v = 0; v < 8; ++v) {
            float lg = acc[t][v] * scaleF;
            if (c != tg[v]) m[v] = fmaxf(m[v], lg);
        }
    }
#pragma unroll
    for (int t = 0; t < 16; ++t) {
        int c = cidx[t * 16 + mrow];
#pragma unroll
        for (int v = 0; v < 8; ++v) {
            float lg = acc[t][v] * scaleF;
            if (c != tg[v] && m[v] > -INFINITY) s[v] += __expf(lg - m[v]);
        }
    }
#pragma unroll
    for (int v = 0; v < 8; ++v) {
#pragma unroll
        for (int off = 1; off < 16; off <<= 1) {
            float mo = __shfl_xor(m[v], off, 32);
            float so = __shfl_xor(s[v], off, 32);
            float M = fmaxf(m[v], mo);
            float S = 0.f;
            if (m[v] > -INFINITY) S += s[v] * __expf(m[v] - M);
            if (mo   > -INFINITY) S += so * __expf(mo - M);
            m[v] = M; s[v] = S;
        }
    }
    if (mrow == 0) {
#pragma unroll
        for (int v = 0; v < 8; ++v) {
            int row = v + 8 * hf;
            out[(size_t)row * NSEG] = make_float2(m[v], s[v]);
        }
    }
}

// ---------------- kernel D: WMMA sims (main + aux) + partial LSE ------------
__global__ __launch_bounds__(256) void k_sims(const _Float16* __restrict__ eF16,
                                              const _Float16* __restrict__ eL16,
                                              const _Float16* __restrict__ hF16,
                                              const _Float16* __restrict__ hL16,
                                              const int* __restrict__ topIdx,
                                              const int* __restrict__ tgt,
                                              const float* __restrict__ pScale,
                                              float2* __restrict__ pm,
                                              float2* __restrict__ pa) {
    __shared__ _Float16 ldsB[SEGC * BSTR];
    __shared__ _Float16 ldsL[SEGC * LSTR];
    __shared__ int cidx[SEGC];

    int seg = blockIdx.x, n = blockIdx.y;
    int tid = threadIdx.x;
    int w = tid >> 5, lane = tid & 31;
    int mrow = lane & 15, hf = lane >> 4;

    float ls = pScale[0];
    float scaleF = 100.f / (1.f + __expf(-ls)) + 1.f;

    int myc = topIdx[n * NCAND + seg * SEGC + tid];
    cidx[tid] = myc;
    const _Float16* myrow = eF16 + (size_t)myc * D_SZ;

    v8f zero = {0.f, 0.f, 0.f, 0.f, 0.f, 0.f, 0.f, 0.f};
    v8f acc[16];
#pragma unroll
    for (int t = 0; t < 16; ++t) acc[t] = zero;

    const int rowBase = n * CHUNK_SZ;
#pragma unroll 1
    for (int kb = 0; kb < D_SZ; kb += KBLK) {
        __syncthreads();
        // stage 256 pre-normalized f16 candidate rows x KBLK cols into LDS
#if HAVE_ASYNC
        {
            const _Float16* g = myrow + kb;
            _Float16* l = &ldsB[tid * BSTR];
#pragma unroll
            for (int j = 0; j < KBLK; j += 8)
                async_cp16(g + j, l + j);
            wait_async0();
        }
#else
        {
            const v8h* g = (const v8h*)(myrow + kb);
            v8h* l = (v8h*)&ldsB[tid * BSTR];
#pragma unroll
            for (int j = 0; j < KBLK / 8; ++j) l[j] = g[j];
        }
#endif
        __syncthreads();
#pragma unroll
        for (int k0 = 0; k0 < KBLK; k0 += 32) {
            const _Float16* ap = hF16 + (size_t)(rowBase + 16 * w + mrow) * D_SZ
                                      + kb + k0 + 8 * hf;
            v16h a = load_a(ap);
#pragma unroll
            for (int t = 0; t < 16; ++t) {
                const _Float16* bp = &ldsB[(t * 16 + mrow) * BSTR + k0 + 8 * hf];
                v16h b = load_b(bp);
                acc[t] = __builtin_amdgcn_wmma_f32_16x16x32_f16(
                    false, a, false, b, (short)0, acc[t], false, false);
            }
        }
    }
    epilogue(acc, cidx, tgt + n * CHUNK_SZ + 16 * w, scaleF,
             pm + (size_t)(n * CHUNK_SZ + 16 * w) * NSEG + seg);

    // ---------- aux (low-rank 32) pass ----------
    __syncthreads();
#if HAVE_ASYNC
    {
        const _Float16* g = eL16 + (size_t)myc * LOWR;
        _Float16* l = &ldsL[tid * LSTR];
#pragma unroll
        for (int j = 0; j < LOWR; j += 8)
            async_cp16(g + j, l + j);
        wait_async0();
    }
#else
    {
        const v8h* g = (const v8h*)(eL16 + (size_t)myc * LOWR);
        v8h* l = (v8h*)&ldsL[tid * LSTR];
#pragma unroll
        for (int j = 0; j < LOWR / 8; ++j) l[j] = g[j];
    }
#endif
    __syncthreads();
#pragma unroll
    for (int t = 0; t < 16; ++t) acc[t] = zero;
    {
        const _Float16* ap = hL16 + (size_t)(rowBase + 16 * w + mrow) * LOWR + 8 * hf;
        v16h a = load_a(ap);
#pragma unroll
        for (int t = 0; t < 16; ++t) {
            const _Float16* bp = &ldsL[(t * 16 + mrow) * LSTR + 8 * hf];
            v16h b = load_b(bp);
            acc[t] = __builtin_amdgcn_wmma_f32_16x16x32_f16(
                false, a, false, b, (short)0, acc[t], false, false);
        }
    }
    epilogue(acc, cidx, tgt + n * CHUNK_SZ + 16 * w, scaleF,
             pa + (size_t)(n * CHUNK_SZ + 16 * w) * NSEG + seg);
}

// ---------------- kernel F: merge partials + target, reduce loss ------------
__device__ __forceinline__ float merge_lse(const float2* p, float tlogit) {
    float M = -INFINITY, S = 0.f;
#pragma unroll
    for (int si = 0; si < NSEG; ++si) {
        float2 q = p[si];
        if (q.x > -INFINITY) {
            if (q.x > M) { S = S * __expf(M - q.x) + q.y; M = q.x; }
            else S += q.y * __expf(q.x - M);
        }
    }
    if (tlogit > M) { S = S * __expf(M - tlogit) + 1.f; M = tlogit; }
    else S += __expf(tlogit - M);
    return logf(S) + M - tlogit;
}

__global__ __launch_bounds__(256) void k_loss(const float2* __restrict__ pm,
                                              const float2* __restrict__ pa,
                                              const float* __restrict__ tF,
                                              const float* __restrict__ tL,
                                              const float* __restrict__ pScale,
                                              float* __restrict__ out) {
    int i = blockIdx.x * blockDim.x + threadIdx.x;
    float ls = pScale[0];
    float scaleF = 100.f / (1.f + __expf(-ls)) + 1.f;
    float contrib = 0.f;
    if (i < T_SZ) {
        float lmain = merge_lse(pm + (size_t)i * NSEG, tF[i] * scaleF);
        float laux  = merge_lse(pa + (size_t)i * NSEG, tL[i] * scaleF);
        contrib = (lmain + 0.2f * laux) / (float)T_SZ;
    }
#pragma unroll
    for (int off = 16; off; off >>= 1) contrib += __shfl_xor(contrib, off, 32);
    if ((threadIdx.x & 31) == 0) atomicAdd(out, contrib);
}

// ---------------- host launcher ---------------------------------------------
extern "C" void kernel_launch(void* const* d_in, const int* in_sizes, int n_in,
                              void* d_out, int out_size, void* d_ws, size_t ws_size,
                              hipStream_t stream) {
    (void)in_sizes; (void)n_in; (void)ws_size;
    const float* h      = (const float*)d_in[0];
    const float* emb    = (const float*)d_in[1];
    const float* lscale = (const float*)d_in[2];
    const int*   tgt    = (const int*)d_in[3];
    float* outp = (float*)d_out;

    char* p = (char*)d_ws;
    auto carve = [&](size_t bytes) -> char* {
        char* r = p;
        p += (bytes + 255) & ~(size_t)255;
        return r;
    };
    float*     invF   = (float*)carve((size_t)V_PAD * 4);
    float*     invL   = (float*)carve((size_t)V_PAD * 4);
    float*     invHF  = (float*)carve((size_t)T_SZ * 4);
    float*     invHL  = (float*)carve((size_t)T_SZ * 4);
    float*     hmean  = (float*)carve((size_t)NCHUNK * LOWR * 4);
    float*     tF     = (float*)carve((size_t)T_SZ * 4);
    float*     tL     = (float*)carve((size_t)T_SZ * 4);
    float*     scores = (float*)carve((size_t)NCHUNK * V_PAD * 4);
    int*       topIdx = (int*)carve((size_t)NCHUNK * NCAND * 4);
    _Float16*  hF16   = (_Float16*)carve((size_t)T_SZ * D_SZ * 2);
    _Float16*  hL16   = (_Float16*)carve((size_t)T_SZ * LOWR * 2);
    float2*    pm     = (float2*)carve((size_t)T_SZ * NSEG * sizeof(float2));
    float2*    pa     = (float2*)carve((size_t)T_SZ * NSEG * sizeof(float2));
    _Float16*  eF16   = (_Float16*)carve((size_t)V_PAD * D_SZ * 2);
    _Float16*  eL16   = (_Float16*)carve((size_t)V_PAD * LOWR * 2);

    (void)hipMemsetAsync(hmean, 0, (size_t)NCHUNK * LOWR * 4, stream);
    (void)hipMemsetAsync(outp, 0, (size_t)out_size * sizeof(float), stream);

    k_emb_norms<<<dim3((V_SZ + 7) / 8), 256, 0, stream>>>(emb, invF, invL, eF16, eL16);
    k_h_norms<<<dim3(T_SZ / 8), 256, 0, stream>>>(h, invHF, invHL, hF16, hL16, hmean);
    k_scan<<<dim3((V_SZ + 255) / 256), 256, 0, stream>>>(eL16, hmean, scores);
    k_topk<<<dim3(NCHUNK), 1024, 0, stream>>>(scores, topIdx);
    k_tgt<<<dim3(T_SZ / 8), 256, 0, stream>>>(h, emb, tgt, invF, invL, invHF, invHL, tF, tL);
    k_sims<<<dim3(NSEG, NCHUNK), 256, 0, stream>>>(eF16, eL16, hF16, hL16,
                                                   topIdx, tgt, lscale, pm, pa);
    k_loss<<<dim3(T_SZ / 256), 256, 0, stream>>>(pm, pa, tF, tL, lscale, outp);
}
